// ConvHydra_35914516529572
// MI455X (gfx1250) — compile-verified, matrix operand order checked by reading, not run on hardware
//
#include <hip/hip_runtime.h>
#include <hip/hip_bf16.h>
#include <cstdint>
#include <cstddef>

// ---------------------------------------------------------------------------
// ConvHydra for MI455X (gfx1250, wave32, WMMA + TDM)
//   conv_in  : implicit GEMM, f16 WMMA, TDM weight staging   (~155 GFLOP)
//   in_proj  : GEMM, f16 WMMA                                (~5.3 GFLOP)
//   dwconv/dt/scan/combine : VALU kernels
//   out_proj : GEMM, f16 WMMA                                (~2.4 GFLOP)
//   conv_out : implicit GEMM, f16 WMMA, TDM weight staging   (~155 GFLOP)
// f16 conv weights (75.5MB each) are L2-resident (192MB); TDM double-buffers
// weight slabs into LDS (hardware padding = bank-conflict-free stride 40)
// while all four waves run WMMA. Activation B-tiles are Hankel: one 96-half
// window per K-slab instead of a 32x64 tile.
// ---------------------------------------------------------------------------

typedef __attribute__((ext_vector_type(16))) _Float16 v16h;
typedef __attribute__((ext_vector_type(8)))  _Float16 v8h;
typedef __attribute__((ext_vector_type(8)))  float    v8f;
typedef __attribute__((ext_vector_type(4)))  unsigned int v4u;
typedef __attribute__((ext_vector_type(8)))  int      v8i;
typedef __attribute__((ext_vector_type(4)))  int      v4i;

#define BB      2
#define SS      1024
#define DD      192
#define KCONV   128
#define DINNER  384
#define PDIM    48
#define NSTATE  32
#define DXBC    448
#define DPROJ   848
#define GG      16      // B*NH sequences entering hydra

#define SA_STRIDE 40    // padded LDS row stride (halfs): 32 data + 8 pad
#define SB_STRIDE 68    // padded LDS row stride (halfs) for generic GEMM B

#if __has_builtin(__builtin_amdgcn_tensor_load_to_lds) && \
    __has_builtin(__builtin_amdgcn_s_wait_tensorcnt)
#define HAVE_TDM 1
#else
#define HAVE_TDM 0
#endif

__device__ __forceinline__ float silu_f(float v) { return v / (1.f + expf(-v)); }

// --------------------------- WMMA fragment loaders --------------------------
// A: 16(M) x 32(K) f16 in LDS, row stride rs.
// wave32: lanes 0-15 -> M=lane, K-phase 0; lanes 16-31 -> M=lane-16, phase 1.
// VGPR i (i<4): K = 8*ph + 2i..2i+1 ; VGPR 4+i: K = 16 + 8*ph + 2i..2i+1.
__device__ __forceinline__ v16h frag_a(const _Float16* p, int rs, int lane) {
  const int ph = (lane >> 4) & 1;
  const int m  = lane & 15;
  const _Float16* r = p + m * rs;
  v16h a;
#pragma unroll
  for (int i = 0; i < 4; ++i) {
    const int k0 = 8 * ph + 2 * i;
    const int k1 = 16 + 8 * ph + 2 * i;
    a[2 * i]         = r[k0];
    a[2 * i + 1]     = r[k0 + 1];
    a[8 + 2 * i]     = r[k1];
    a[8 + 2 * i + 1] = r[k1 + 1];
  }
  return a;
}

// B: 32(K) x 16(N), element (k,n) = p[k*rs + n]. rs==1 gives Hankel reads.
__device__ __forceinline__ v16h frag_b(const _Float16* p, int rs, int lane) {
  const int ph = (lane >> 4) & 1;
  const int n  = lane & 15;
  v16h b;
#pragma unroll
  for (int i = 0; i < 4; ++i) {
    const int k0 = 8 * ph + 2 * i;
    const int k1 = 16 + 8 * ph + 2 * i;
    b[2 * i]         = p[k0 * rs + n];
    b[2 * i + 1]     = p[(k0 + 1) * rs + n];
    b[8 + 2 * i]     = p[k1 * rs + n];
    b[8 + 2 * i + 1] = p[(k1 + 1) * rs + n];
  }
  return b;
}

#define WMMA_F16(A_, B_, C_) \
  __builtin_amdgcn_wmma_f32_16x16x32_f16(false, (A_), false, (B_), (short)0, (C_), false, false)

// --------------------------- TDM weight-slab load ---------------------------
// 2D D# per CDNA5 ISA:
//   tile: rows x 32 halfs, row stride = row_stride elems, data_size = 2B,
//   LDS pad: after every 16 DWORDs (32 halfs) insert 4 DWORDs -> stride 40.
// This toolchain exposes the 6-arg builtin:
//   (uint32x4 g0, int32x8 g1, int32x4 g2, int32x4 g3, int32x8 g4, i32 cpol)
__device__ __forceinline__ void tdm_load_a(const _Float16* gptr, unsigned lds_off,
                                           int rows, long long row_stride) {
#if HAVE_TDM
  const unsigned long long ga = (unsigned long long)(uintptr_t)gptr;
  v4u g0;
  g0[0] = 1u;                                           // count=1, user D#
  g0[1] = lds_off;                                      // lds_addr (bytes)
  g0[2] = (unsigned)(ga & 0xffffffffu);                 // global_addr[31:0]
  g0[3] = (unsigned)((ga >> 32) & 0x01ffffffu)          // global_addr[56:32]
        | (2u << 30);                                   // type=2 (image)
  v8i g1;
  g1[0] = (1 << 16)                                     // data_size = 2 bytes
        | (1 << 20)                                     // pad_enable
        | (3 << 22)                                     // pad_interval: 16 DW
        | (3 << 25);                                    // pad_amount: 4 DW
  g1[1] = (32 << 16);                                   // tensor_dim0 = 32
  g1[2] = (rows << 16);                                 // tensor_dim1 = rows
  g1[3] = (32 << 16);                                   // tile_dim0 = 32
  g1[4] = rows;                                         // tile_dim1 = rows
  g1[5] = (int)(row_stride & 0xffffffffll);             // dim0_stride[31:0]
  g1[6] = (int)((row_stride >> 32) & 0xffffll);         // dim0_stride[47:32]
  g1[7] = 0;
  v4i g2 = {0, 0, 0, 0};                                // 2D tensor: groups 2/3 unused
  v4i g3 = {0, 0, 0, 0};
  v8i g4 = {0, 0, 0, 0, 0, 0, 0, 0};
  __builtin_amdgcn_tensor_load_to_lds(g0, g1, g2, g3, g4, 0);
#else
  (void)gptr; (void)lds_off; (void)rows; (void)row_stride;
#endif
}

// --------------------------- f32 -> f16 convert -----------------------------
__global__ void cvt_kernel(const float* __restrict__ s, _Float16* __restrict__ d, int n) {
  int i = blockIdx.x * blockDim.x + threadIdx.x;
  if (i < n) d[i] = (_Float16)s[i];
}

// --------------------------- implicit-GEMM conv1d ---------------------------
// out[b][co][t] = silu( bias[co] + sum_{ci,k} X[b][ci][t+k-63] * W[co][ci][k] )
// Block: 128 threads (4 waves, 2x2), tile BM(co) x 64(t), K-slab 32.
// Weight slab via TDM (double-buffered); activation slab = 96-half window.
template<int CIN, int COUT, int BM, bool WRITE_H>
__global__ void __launch_bounds__(128)
conv_wmma_kernel(const _Float16* __restrict__ Wh,   // [COUT][CIN][128] f16
                 const _Float16* __restrict__ Xh,   // [B][CIN][1024]  f16
                 const float*    __restrict__ bias, // [COUT]
                 float*          __restrict__ outF, // [B][COUT][1024] (!WRITE_H)
                 _Float16*       __restrict__ outH) // same flat layout (WRITE_H)
{
  constexpr int MF = BM / 32;                 // M-frags per wave (wave: BM/2 x 32)
  __shared__ _Float16 sA[2][BM * SA_STRIDE];
  __shared__ _Float16 win[2][128];            // 96 used (+ slack)

  const int tid  = threadIdx.x;
  const int lane = tid & 31;
  const int wave = tid >> 5;
  const int wm   = wave >> 1;
  const int wn   = wave & 1;
  const int co0  = blockIdx.x * BM;
  const int t0   = blockIdx.y * 64;
  const int bb   = blockIdx.z;
  const size_t xbase = (size_t)bb * CIN * SS;
  const size_t wbase = (size_t)co0 * CIN * KCONV;   // row m slab k0 at wbase + m*CIN*128 + k0

  v8f acc[MF][2] = {};

  const int Ktot = CIN * KCONV;

  auto stage = [&](int b, int k0) {
    // ---- weights: BM x 32 slab ----
#if HAVE_TDM
    if (wave == 0)
      tdm_load_a(Wh + wbase + (size_t)k0, (unsigned)(uintptr_t)&sA[b][0],
                 BM, (long long)CIN * KCONV);
#else
    for (int c = tid; c < (BM * 32) / 8; c += 128) {
      const int flat = c * 8;
      const int m  = flat >> 5;
      const int kk = flat & 31;
      v8h wv = *(const v8h*)(Wh + wbase + (size_t)m * CIN * KCONV + k0 + kk);
      _Float16* dst = &sA[b][m * SA_STRIDE + kk];
#pragma unroll
      for (int j = 0; j < 8; ++j) dst[j] = wv[j];
    }
#endif
    // ---- activations: Hankel window win[j] = X[ci, t0-63+ko0+j], j<96 ----
    const int ci  = k0 >> 7;
    const int ko0 = k0 & 127;
    for (int j = tid; j < 96; j += 128) {
      const int tp = t0 - 63 + ko0 + j;           // 'same' pad: lp = 63
      _Float16 v = (_Float16)0.f;
      if (tp >= 0 && tp < SS) v = Xh[xbase + (size_t)ci * SS + tp];
      win[b][j] = v;
    }
  };

  // prologue
  stage(0, 0);
#if HAVE_TDM
  if (wave == 0) __builtin_amdgcn_s_wait_tensorcnt(0);
#endif
  __syncthreads();

  int buf = 0;
  for (int k0 = 0; k0 < Ktot; k0 += 32) {
    const int kn = k0 + 32;
    if (kn < Ktot) stage(buf ^ 1, kn);            // overlap next slab

    const v16h b0 = frag_b(&win[buf][wn * 32 + 0],  1, lane);
    const v16h b1 = frag_b(&win[buf][wn * 32 + 16], 1, lane);
#pragma unroll
    for (int mi = 0; mi < MF; ++mi) {
      const v16h a = frag_a(&sA[buf][(wm * (BM / 2) + mi * 16) * SA_STRIDE],
                            SA_STRIDE, lane);
      acc[mi][0] = WMMA_F16(a, b0, acc[mi][0]);
      acc[mi][1] = WMMA_F16(a, b1, acc[mi][1]);
    }
#if HAVE_TDM
    if (wave == 0 && kn < Ktot) __builtin_amdgcn_s_wait_tensorcnt(0);
#endif
    __syncthreads();
    buf ^= 1;
  }

  // Epilogue: C/D layout -> VGPR i holds M = i + 8*(lane>=16), N = lane&15.
  const int ph = lane >> 4;
  const int nn = lane & 15;
#pragma unroll
  for (int mi = 0; mi < MF; ++mi)
#pragma unroll
    for (int ni = 0; ni < 2; ++ni)
#pragma unroll
      for (int i = 0; i < 8; ++i) {
        const int co = co0 + wm * (BM / 2) + mi * 16 + i + 8 * ph;
        const int t  = t0  + wn * 32 + ni * 16 + nn;
        float v = acc[mi][ni][i] + bias[co];
        v = silu_f(v);
        const size_t o = ((size_t)bb * COUT + co) * SS + t;
        if (WRITE_H) outH[o] = (_Float16)v;
        else         outF[o] = v;
      }
}

// --------------------------- generic GEMM (f16 in, f32 acc) -----------------
// C[m][n] = sum_k A[m][k] * B[k][n];  A: MxK row-major, B: KxN row-major.
// M % 64 == 0, K % 32 == 0; N bounds-checked.
template<bool WRITE_H>
__global__ void __launch_bounds__(128)
gemm_wmma_kernel(const _Float16* __restrict__ A,
                 const _Float16* __restrict__ Bm,
                 float*          __restrict__ Cf,
                 _Float16*       __restrict__ Ch,
                 int M, int N, int Kd)
{
  __shared__ _Float16 sA[64 * SA_STRIDE];
  __shared__ _Float16 sB[32 * SB_STRIDE];

  const int tid  = threadIdx.x;
  const int lane = tid & 31;
  const int wave = tid >> 5;
  const int wm   = wave >> 1;
  const int wn   = wave & 1;
  const int m0   = blockIdx.x * 64;
  const int n0   = blockIdx.y * 64;

  v8f acc[2][2] = {};

  for (int k0 = 0; k0 < Kd; k0 += 32) {
    for (int c = tid; c < 512; c += 128) {
      const int flat = c * 8;
      const int m  = flat >> 5;
      const int kk = flat & 31;
      v8h av = *(const v8h*)(A + (size_t)(m0 + m) * Kd + k0 + kk);
      _Float16* dst = &sA[m * SA_STRIDE + kk];
#pragma unroll
      for (int j = 0; j < 8; ++j) dst[j] = av[j];
    }
    for (int e = tid; e < 2048; e += 128) {
      const int kk = e >> 6;
      const int n  = e & 63;
      const int gn = n0 + n;
      _Float16 v = (_Float16)0.f;
      if (gn < N) v = Bm[(size_t)(k0 + kk) * N + gn];
      sB[kk * SB_STRIDE + n] = v;
    }
    __syncthreads();

    if (k0 + 32 < Kd)
      __builtin_prefetch((const void*)(A + (size_t)(m0 + (tid >> 1)) * Kd + k0 + 32), 0, 2);

    const v16h a0 = frag_a(&sA[(wm * 32 + 0)  * SA_STRIDE], SA_STRIDE, lane);
    const v16h a1 = frag_a(&sA[(wm * 32 + 16) * SA_STRIDE], SA_STRIDE, lane);
    const v16h b0 = frag_b(&sB[wn * 32 + 0],  SB_STRIDE, lane);
    const v16h b1 = frag_b(&sB[wn * 32 + 16], SB_STRIDE, lane);
    acc[0][0] = WMMA_F16(a0, b0, acc[0][0]);
    acc[0][1] = WMMA_F16(a0, b1, acc[0][1]);
    acc[1][0] = WMMA_F16(a1, b0, acc[1][0]);
    acc[1][1] = WMMA_F16(a1, b1, acc[1][1]);
    __syncthreads();
  }

  const int ph = lane >> 4;
  const int nn = lane & 15;
#pragma unroll
  for (int mi = 0; mi < 2; ++mi)
#pragma unroll
    for (int ni = 0; ni < 2; ++ni)
#pragma unroll
      for (int i = 0; i < 8; ++i) {
        const int m = m0 + wm * 32 + mi * 16 + i + 8 * ph;
        const int n = n0 + wn * 32 + ni * 16 + nn;
        if (n < N) {
          const float v = acc[mi][ni][i];
          if (WRITE_H) Ch[(size_t)m * N + n] = (_Float16)v;
          else         Cf[(size_t)m * N + n] = v;
        }
      }
}

// --------------------------- depthwise conv (k=7) + silu --------------------
__global__ void dwconv_silu_kernel(const float* __restrict__ zx,    // [GG][SS][848]
                                   const float* __restrict__ cw,    // [448][1][7]
                                   const float* __restrict__ cb,    // [448]
                                   float*       __restrict__ out)   // [GG][SS][448]
{
  const int idx = blockIdx.x * blockDim.x + threadIdx.x;
  if (idx >= GG * SS * DXBC) return;
  const int c = idx % DXBC;
  const int t = (idx / DXBC) % SS;
  const int g = idx / (DXBC * SS);
  float acc = cb[c];
#pragma unroll
  for (int j = 0; j < 7; ++j) {
    const int tt = t + j - 3;                       // 'same' pad: lp = 3
    if (tt >= 0 && tt < SS)
      acc += zx[((size_t)g * SS + tt) * DPROJ + DINNER + c] * cw[c * 7 + j];
  }
  out[idx] = silu_f(acc);
}

// --------------------------- dt = softplus(... + bias) ----------------------
__global__ void dt_softplus_kernel(const float* __restrict__ zx,
                                   const float* __restrict__ dtb,
                                   float*       __restrict__ dt)
{
  const int idx = blockIdx.x * blockDim.x + threadIdx.x;
  if (idx >= GG * SS * 16) return;
  const int i  = idx & 15;
  const int gt = idx >> 4;
  const float v = zx[(size_t)gt * DPROJ + (DINNER + DXBC) + i] + dtb[i];
  dt[idx] = (v > 20.f) ? v : log1pf(expf(v));
}

// --------------------------- bidirectional SSD scan -------------------------
// One block per (dir, seq, head); thread p (<48) owns state row p (32 regs).
__global__ void __launch_bounds__(64)
ssd_scan_kernel(const float* __restrict__ xbc,   // [GG][SS][448]
                const float* __restrict__ dt,    // [GG][SS][16]
                const float* __restrict__ A_log, // [16]
                float* __restrict__ y_fw,        // [GG][SS][8][48]
                float* __restrict__ y_bw)        // [GG][SS][8][48]
{
  const int bidx = blockIdx.x;          // 0..255
  const int dir  = bidx >> 7;
  const int g    = (bidx >> 3) & 15;
  const int h    = bidx & 7;
  const int tid  = threadIdx.x;

  const float A = -expf(A_log[dir * 8 + h]);
  float state[NSTATE];
#pragma unroll
  for (int n = 0; n < NSTATE; ++n) state[n] = 0.f;

  __shared__ float sBC[64];             // B[0..31], C[32..63]
  float* __restrict__ yout = dir ? y_bw : y_fw;

  for (int step = 0; step < SS; ++step) {
    const int t = dir ? (SS - 1 - step) : step;
    const float* row = xbc + ((size_t)g * SS + t) * DXBC;
    sBC[tid] = row[DINNER + tid];       // 64 threads cover B and C
    const float dtv = dt[((size_t)g * SS + t) * 16 + dir * 8 + h];
    __syncthreads();
    if (tid < PDIM) {
      const float xd  = row[h * PDIM + tid] * dtv;
      const float dec = __expf(dtv * A);
      float y = 0.f;
#pragma unroll
      for (int n = 0; n < NSTATE; ++n) {
        state[n] = state[n] * dec + xd * sBC[n];
        y += state[n] * sBC[NSTATE + n];
      }
      yout[(((size_t)g * SS + t) * 8 + h) * PDIM + tid] = y;
    }
    __syncthreads();
  }
}

// ------------------- shift + gate + RMSNorm -> f16 rows ---------------------
// One wave per (g,t) row of 384; __shfl_xor wave32 reduction.
__global__ void __launch_bounds__(128)
combine_rmsnorm_kernel(const float* __restrict__ y_fw,
                       const float* __restrict__ y_bw,
                       const float* __restrict__ xbc,
                       const float* __restrict__ zx,
                       const float* __restrict__ Dp,
                       const float* __restrict__ norm_w,
                       _Float16*    __restrict__ y_h)   // [GG*SS][384]
{
  const int lane = threadIdx.x & 31;
  const int wave = threadIdx.x >> 5;
  const int row  = blockIdx.x * 4 + wave;   // 0..16383
  const int g = row >> 10;
  const int t = row & 1023;

  float vals[12];
  float ss = 0.f;
#pragma unroll
  for (int j = 0; j < 12; ++j) {
    const int c = j * 32 + lane;
    const int h = c / PDIM;
    const int p = c - h * PDIM;
    const float yf = (t > 0)      ? y_fw[(((size_t)g * SS + (t - 1)) * 8 + h) * PDIM + p] : 0.f;
    const float yb = (t < SS - 1) ? y_bw[(((size_t)g * SS + (t + 1)) * 8 + h) * PDIM + p] : 0.f;
    const float x  = xbc[((size_t)g * SS + t) * DXBC + c];
    const float z  = zx[((size_t)g * SS + t) * DPROJ + c];
    float y = (yf + yb + x * Dp[h]) * silu_f(z);
    vals[j] = y;
    ss += y * y;
  }
#pragma unroll
  for (int o = 16; o > 0; o >>= 1) ss += __shfl_xor(ss, o, 32);
  const float r = rsqrtf(ss / (float)DINNER + 1e-5f);
#pragma unroll
  for (int j = 0; j < 12; ++j) {
    const int c = j * 32 + lane;
    y_h[(size_t)row * DINNER + c] = (_Float16)(vals[j] * r * norm_w[c]);
  }
}

// ---------------------------------------------------------------------------
extern "C" void kernel_launch(void* const* d_in, const int* in_sizes, int n_in,
                              void* d_out, int out_size, void* d_ws, size_t ws_size,
                              hipStream_t stream) {
  (void)in_sizes; (void)n_in; (void)out_size; (void)ws_size;

  const float* x       = (const float*)d_in[0];
  const float* Wc_in   = (const float*)d_in[1];
  const float* bc_in   = (const float*)d_in[2];
  const float* Wc_out  = (const float*)d_in[3];
  const float* bc_out  = (const float*)d_in[4];
  const float* W_in    = (const float*)d_in[5];
  const float* conv_w  = (const float*)d_in[6];
  const float* conv_b  = (const float*)d_in[7];
  const float* dt_bias = (const float*)d_in[8];
  const float* A_log   = (const float*)d_in[9];
  const float* Dp      = (const float*)d_in[10];
  const float* norm_w  = (const float*)d_in[11];
  const float* W_out   = (const float*)d_in[12];
  float* out = (float*)d_out;

  char* ws = (char*)d_ws;
  size_t off = 0;
  auto alloc = [&](size_t bytes) -> char* {
    char* p = ws + off;
    off += (bytes + 255) & ~(size_t)255;
    return p;
  };

  _Float16* wc_in_h  = (_Float16*)alloc((size_t)1536 * 192 * 128 * 2);
  _Float16* wc_out_h = (_Float16*)alloc((size_t)192 * 1536 * 128 * 2);
  _Float16* x_h      = (_Float16*)alloc((size_t)BB * DD * SS * 2);
  _Float16* w_in_h   = (_Float16*)alloc((size_t)DD * DPROJ * 2);
  _Float16* w_out_h  = (_Float16*)alloc((size_t)DINNER * DD * 2);
  _Float16* u_h      = (_Float16*)alloc((size_t)GG * SS * DD * 2);
  float*    zx       = (float*)   alloc((size_t)GG * SS * DPROJ * 4);
  float*    xbc      = (float*)   alloc((size_t)GG * SS * DXBC * 4);
  float*    dt       = (float*)   alloc((size_t)GG * SS * 16 * 4);
  float*    yfw      = (float*)   alloc((size_t)GG * SS * DINNER * 4);
  float*    ybw      = (float*)   alloc((size_t)GG * SS * DINNER * 4);
  _Float16* y_h      = (_Float16*)alloc((size_t)GG * SS * DINNER * 2);
  _Float16* xo_h     = (_Float16*)alloc((size_t)GG * SS * DD * 2);

  auto cvt = [&](const float* s, _Float16* d, size_t n) {
    cvt_kernel<<<(unsigned)((n + 255) / 256), 256, 0, stream>>>(s, d, (int)n);
  };
  cvt(Wc_in,  wc_in_h,  (size_t)1536 * 192 * 128);
  cvt(Wc_out, wc_out_h, (size_t)192 * 1536 * 128);
  cvt(x,      x_h,      (size_t)BB * DD * SS);
  cvt(W_in,   w_in_h,   (size_t)DD * DPROJ);
  cvt(W_out,  w_out_h,  (size_t)DINNER * DD);

  // conv_in: (B,192,1024) -> silu -> u_h as (16,1024,192) flat. BM=128.
  conv_wmma_kernel<192, 1536, 128, true>
      <<<dim3(1536 / 128, SS / 64, BB), 128, 0, stream>>>(wc_in_h, x_h, bc_in, nullptr, u_h);

  // in_proj: (16384 x 192) @ (192 x 848) -> zx (f32)
  gemm_wmma_kernel<false>
      <<<dim3((GG * SS) / 64, (DPROJ + 63) / 64), 128, 0, stream>>>(
          u_h, w_in_h, zx, nullptr, GG * SS, DPROJ, DD);

  dwconv_silu_kernel<<<(GG * SS * DXBC + 255) / 256, 256, 0, stream>>>(zx, conv_w, conv_b, xbc);
  dt_softplus_kernel<<<(GG * SS * 16 + 255) / 256, 256, 0, stream>>>(zx, dt_bias, dt);

  // 256 independent scans: (2 dirs x 16 seqs x 8 heads), 64 threads each
  ssd_scan_kernel<<<256, 64, 0, stream>>>(xbc, dt, A_log, yfw, ybw);

  combine_rmsnorm_kernel<<<(GG * SS) / 4, 128, 0, stream>>>(yfw, ybw, xbc, zx, Dp, norm_w, y_h);

  // out_proj: (16384 x 384) @ (384 x 192) -> xo_h (f16), flat == (B,1536,1024)
  gemm_wmma_kernel<true>
      <<<dim3((GG * SS) / 64, DD / 64), 128, 0, stream>>>(
          y_h, w_out_h, nullptr, xo_h, GG * SS, DD, DINNER);

  // conv_out: (B,1536,1024) -> silu -> d_out (f32, flat == (B,1024,192)). BM=64.
  conv_wmma_kernel<1536, 192, 64, false>
      <<<dim3(192 / 64, SS / 64, BB), 128, 0, stream>>>(wc_out_h, xo_h, bc_out, out, nullptr);
}
